// Model_31241592111127
// MI455X (gfx1250) — matrix operand order repaired and checked by reading, not compiled, not run on hardware
//
#include <hip/hip_runtime.h>
#include <math.h>

typedef __attribute__((ext_vector_type(16))) __bf16        bf16x16;
typedef __attribute__((ext_vector_type(2)))  __bf16        bf16x2;
typedef __attribute__((ext_vector_type(8)))  float         f32x8;
typedef __attribute__((ext_vector_type(4)))  unsigned int  u32x4;
typedef unsigned short u16;

#define DEVI __device__ __forceinline__

union BF2 { bf16x2 v; unsigned u; };
union BFS { __bf16 h; u16 u; };
union Frag { u32x4 q[2]; bf16x16 v; };
union F8   { f32x8 v; float f[8]; };

DEVI unsigned pack_bf16(float a, float b) {   // hardware cvt, RNE
    BF2 p;
    p.v[0] = (__bf16)a;
    p.v[1] = (__bf16)b;
    return p.u;
}
DEVI u16 f2bf16(float a) { BFS s; s.h = (__bf16)a; return s.u; }
DEVI float bf2f(u16 x) { BFS s; s.u = x; return (float)s.h; }

// A-operand (16x32 bf16): lanes<16 hold K {0..7,16..23}, lanes>=16 hold K {8..15,24..31}
DEVI bf16x16 load_frag_a(const u16* rowp, int hf) {
    int k0 = hf * 8;
    Frag fr;
    fr.q[0] = *(const u32x4*)(rowp + k0);
    fr.q[1] = *(const u32x4*)(rowp + k0 + 16);
    return fr.v;
}
// B-operand (32x16 bf16, K-contiguous per column): lanes<16 hold K {0..7,16..23}, lanes>=16 hold K {8..15,24..31}
DEVI bf16x16 load_frag_b(const u16* rowp, int hf) {
    int k0 = hf * 8;
    Frag fr;
    fr.q[0] = *(const u32x4*)(rowp + k0);
    fr.q[1] = *(const u32x4*)(rowp + k0 + 16);
    return fr.v;
}
DEVI f32x8 wmma_bf16(bf16x16 a, bf16x16 b, f32x8 c) {
    return __builtin_amdgcn_wmma_f32_16x16x32_bf16(false, a, false, b, (short)0, c, false, false);
}
DEVI float sigm(float x) { return 1.f / (1.f + __expf(-x)); }

// ---------------------------------------------------------------- prep kernels

__global__ __launch_bounds__(256) void embed_k(const int* __restrict__ targets,
                                               const float* __restrict__ emb,
                                               float* __restrict__ x) {
    int idx = blockIdx.x * 256 + threadIdx.x;          // 1024*256
    int row = idx >> 8, e = idx & 255;
    x[idx] = emb[(size_t)targets[row] * 256 + e];
}

// B[K,N] f32 row-major -> BT[NP][K] bf16 (zero-padded rows for n >= N).
// grid = NP blocks, block = K/2 threads.
__global__ void btr_k(const float* __restrict__ B, u16* __restrict__ BT, int N, int K) {
    int n = blockIdx.x;
    int k = threadIdx.x * 2;
    float v0 = 0.f, v1 = 0.f;
    if (n < N) { v0 = B[(size_t)k * N + n]; v1 = B[(size_t)(k + 1) * N + n]; }
    ((unsigned*)BT)[((size_t)n * K + k) >> 1] = pack_bf16(v0, v1);
}

__global__ __launch_bounds__(512) void nullstate_k(const int* __restrict__ tnull,
                                                   const float* __restrict__ emb,
                                                   const float* __restrict__ nnW,
                                                   const float* __restrict__ nnb,
                                                   float* __restrict__ ns) {
    int u = threadIdx.x;
    int t = tnull[0];
    const float* xr = emb + (size_t)t * 256;
    float acc = nnb[u];
    for (int e = 0; e < 256; e++) acc += xr[e] * nnW[e * 512 + u];
    ns[u] = tanhf(acc);
}

// ---------------------------------------------------------------- generic WMMA GEMM
// C[M,N] = act(A[M,K] @ B[K,N] + bias). A f32 row-major; B supplied as BT[NP][K] bf16.
// Double-buffered A staging in LDS; B fragments straight from global (L2 resident).
template <int ACT>
__global__ __launch_bounds__(128) void gemm_k(const float* __restrict__ A,
                                              const u16* __restrict__ BT,
                                              const float* __restrict__ bias,
                                              float* __restrict__ C,
                                              int M, int N, int K, int ldc) {
    __shared__ u16 aT[2][64 * 40];
    int tid = threadIdx.x;
    int wv = tid >> 5, lane = tid & 31, hf = lane >> 4, lrow = lane & 15;
    int nbase = blockIdx.x * 64, mbase = blockIdx.y * 64;
    int row = tid >> 1, kc = (tid & 1) * 16;
    int grow = mbase + row;

    f32x8 acc[4];
#pragma unroll
    for (int t = 0; t < 4; t++) {
        int col = nbase + t * 16 + lrow;
        float bvv = (bias && col < N) ? bias[col] : 0.f;
        F8 tmp;
#pragma unroll
        for (int r = 0; r < 8; r++) tmp.f[r] = bvv;
        acc[t] = tmp.v;
    }

    auto stage = [&](u16* dst, int kk0) {
        unsigned pk[8];
        if (grow < M) {
            const float* ap = A + (size_t)grow * K + kk0 + kc;
#pragma unroll
            for (int i = 0; i < 8; i++) pk[i] = pack_bf16(ap[2 * i], ap[2 * i + 1]);
        } else {
#pragma unroll
            for (int i = 0; i < 8; i++) pk[i] = 0u;
        }
        u32x4* d = (u32x4*)(dst + row * 40 + kc);
        d[0] = (u32x4){pk[0], pk[1], pk[2], pk[3]};
        d[1] = (u32x4){pk[4], pk[5], pk[6], pk[7]};
    };

    int nk = K >> 5;
    stage(aT[0], 0);
    __syncthreads();
    for (int kt = 0; kt < nk; kt++) {
        int kk = kt << 5;
        if (kt + 1 < nk) stage(aT[(kt + 1) & 1], kk + 32);
        const u16* ab = aT[kt & 1];
        bf16x16 a = load_frag_a(&ab[(wv * 16 + lrow) * 40], hf);
#pragma unroll
        for (int t = 0; t < 4; t++) {
            int n = nbase + t * 16 + lrow;
            bf16x16 b = load_frag_b(BT + (size_t)n * K + kk, hf);
            acc[t] = wmma_bf16(a, b, acc[t]);
        }
        __syncthreads();
    }

#pragma unroll
    for (int t = 0; t < 4; t++) {
        int col = nbase + t * 16 + lrow;
        if (col >= N) continue;
#pragma unroll
        for (int r = 0; r < 8; r++) {
            int rr = mbase + wv * 16 + hf * 8 + r;
            if (rr < M) {
                float v = acc[t][r];
                if (ACT == 1) v = tanhf(v);
                C[(size_t)rr * ldc + col] = v;
            }
        }
    }
}

// ---------------------------------------------------------------- fused logits + softmax stats
// L = ts[1024,512] @ Wv[512,32000] + bv; per 64x64 block emit per-row (max, sumexp).
__global__ __launch_bounds__(128) void logits_reduce_k(const float* __restrict__ ts,
                                                       const u16* __restrict__ WvT,  // [32000][512] bf16
                                                       const float* __restrict__ bv,
                                                       float2* __restrict__ pb) {
    const int K = 512, NB = 500;
    __shared__ u16 aT[2][64 * 40];
    int tid = threadIdx.x;
    int wv = tid >> 5, lane = tid & 31, hf = lane >> 4, lrow = lane & 15;
    int nbase = blockIdx.x * 64, mbase = blockIdx.y * 64;
    int row = tid >> 1, kc = (tid & 1) * 16;
    int grow = mbase + row;

    f32x8 acc[4];
#pragma unroll
    for (int t = 0; t < 4; t++) {
        float bvv = bv[nbase + t * 16 + lrow];
        F8 tmp;
#pragma unroll
        for (int r = 0; r < 8; r++) tmp.f[r] = bvv;
        acc[t] = tmp.v;
    }

    auto stage = [&](u16* dst, int kk0) {
        const float* ap = ts + (size_t)grow * K + kk0 + kc;
        unsigned pk[8];
#pragma unroll
        for (int i = 0; i < 8; i++) pk[i] = pack_bf16(ap[2 * i], ap[2 * i + 1]);
        u32x4* d = (u32x4*)(dst + row * 40 + kc);
        d[0] = (u32x4){pk[0], pk[1], pk[2], pk[3]};
        d[1] = (u32x4){pk[4], pk[5], pk[6], pk[7]};
    };

    stage(aT[0], 0);
    __syncthreads();
    for (int kt = 0; kt < 16; kt++) {
        int kk = kt << 5;
        if (kt + 1 < 16) stage(aT[(kt + 1) & 1], kk + 32);
        const u16* ab = aT[kt & 1];
        bf16x16 a = load_frag_a(&ab[(wv * 16 + lrow) * 40], hf);
#pragma unroll
        for (int t = 0; t < 4; t++) {
            int n = nbase + t * 16 + lrow;
            bf16x16 b = load_frag_b(WvT + (size_t)n * K + kk, hf);
            acc[t] = wmma_bf16(a, b, acc[t]);
        }
        __syncthreads();
    }

#pragma unroll
    for (int r = 0; r < 8; r++) {
        float m = fmaxf(fmaxf(acc[0][r], acc[1][r]), fmaxf(acc[2][r], acc[3][r]));
        for (int d = 8; d >= 1; d >>= 1) m = fmaxf(m, __shfl_xor(m, d, 32));
        float s = expf(acc[0][r] - m) + expf(acc[1][r] - m) +
                  expf(acc[2][r] - m) + expf(acc[3][r] - m);
        for (int d = 8; d >= 1; d >>= 1) s += __shfl_xor(s, d, 32);
        if (lrow == 0) {
            int rr = mbase + wv * 16 + hf * 8 + r;
            pb[(size_t)rr * NB + blockIdx.x] = make_float2(m, s);
        }
    }
}

__global__ __launch_bounds__(256) void combine_k(const float2* __restrict__ pb,
                                                 float* __restrict__ rowmax,
                                                 float* __restrict__ rowsum) {
    int wv = threadIdx.x >> 5, lane = threadIdx.x & 31;
    int row = blockIdx.x * 8 + wv;
    float m = -1e30f;
    for (int i = lane; i < 500; i += 32) m = fmaxf(m, pb[(size_t)row * 500 + i].x);
    for (int d = 16; d >= 1; d >>= 1) m = fmaxf(m, __shfl_xor(m, d, 32));
    float s = 0.f;
    for (int i = lane; i < 500; i += 32) {
        float2 p = pb[(size_t)row * 500 + i];
        s += p.y * expf(p.x - m);
    }
    for (int d = 16; d >= 1; d >>= 1) s += __shfl_xor(s, d, 32);
    if (lane == 0) { rowmax[row] = m; rowsum[row] = s; }
}

// ---------------------------------------------------------------- bi-LSTM (1 workgroup per direction)
__global__ __launch_bounds__(512) void lstm_k(const float* __restrict__ xg_fw,
                                              const float* __restrict__ xg_bw,
                                              const u16* __restrict__ WhT_fw,
                                              const u16* __restrict__ WhT_bw,
                                              const int* __restrict__ lengths,
                                              float* __restrict__ hcat) {
    int dir = blockIdx.x;
    const float* xg = dir ? xg_bw : xg_fw;
    const u16* Wt = dir ? WhT_bw : WhT_fw;
    __shared__ u16 hl[16 * 520];
    __shared__ int len_s[16];
    __shared__ int tsrc_s[16];
    int tid = threadIdx.x, wv = tid >> 5, lane = tid & 31, hf = lane >> 4, lrow = lane & 15;
    for (int i = tid; i < 16 * 520; i += 512) hl[i] = 0;
    if (tid < 16) len_s[tid] = lengths[tid];
    __syncthreads();

    float c0[8], c1[8];
#pragma unroll
    for (int r = 0; r < 8; r++) { c0[r] = 0.f; c1[r] = 0.f; }

    for (int t = 0; t < 64; t++) {
        if (tid < 16) {
            int L = len_s[tid];
            tsrc_s[tid] = (dir && t < L) ? (L - 1 - t) : t;
        }
        __syncthreads();

        f32x8 acc[4][2];
#pragma unroll
        for (int g = 0; g < 4; g++)
#pragma unroll
            for (int sb = 0; sb < 2; sb++) {
                int col = g * 512 + wv * 32 + sb * 16 + lrow;
                F8 tmp;
#pragma unroll
                for (int r = 0; r < 8; r++) {
                    int b = hf * 8 + r;
                    tmp.f[r] = xg[((size_t)(b * 64 + tsrc_s[b])) * 2048 + col];
                }
                acc[g][sb] = tmp.v;
            }

        for (int kk = 0; kk < 512; kk += 32) {
            bf16x16 a = load_frag_a(&hl[lrow * 520 + kk], hf);
#pragma unroll
            for (int g = 0; g < 4; g++)
#pragma unroll
                for (int sb = 0; sb < 2; sb++) {
                    int n = g * 512 + wv * 32 + sb * 16 + lrow;
                    bf16x16 bfr = load_frag_b(Wt + (size_t)n * 512 + kk, hf);
                    acc[g][sb] = wmma_bf16(a, bfr, acc[g][sb]);
                }
        }
        __syncthreads();

#pragma unroll
        for (int sb = 0; sb < 2; sb++) {
            float* cc = sb ? c1 : c0;
            int u = wv * 32 + sb * 16 + lrow;
#pragma unroll
            for (int r = 0; r < 8; r++) {
                int b = hf * 8 + r;
                bool mk = t < len_s[b];
                float gi = acc[0][sb][r], gj = acc[1][sb][r];
                float gf = acc[2][sb][r], go = acc[3][sb][r];
                float cn = sigm(gf + 1.f) * cc[r] + sigm(gi) * tanhf(gj);
                float hn = sigm(go) * tanhf(cn);
                if (mk) { cc[r] = cn; hl[b * 520 + u] = f2bf16(hn); }
                hcat[((size_t)(b * 64 + tsrc_s[b])) * 1024 + dir * 512 + u] = mk ? hn : 0.f;
            }
        }
        __syncthreads();
    }
}

// ---------------------------------------------------------------- emission path tails

__global__ __launch_bounds__(64) void gl_k(const float* __restrict__ ts,
                                           const u16* __restrict__ WvT,
                                           const float* __restrict__ bv,
                                           const int* __restrict__ sources,
                                           float* __restrict__ gl) {
    __shared__ float tsr[512];
    int row = blockIdx.x;       // b*64+t
    int b = row >> 6;
    for (int i = threadIdx.x; i < 512; i += 64)
        tsr[i] = bf2f(f2bf16(ts[(size_t)row * 512 + i]));   // match GEMM precision
    __syncthreads();
    int s = threadIdx.x;
    int col = sources[b * 64 + s];
    const u16* wp = WvT + (size_t)col * 512;
    float acc = bv[col];
    for (int k = 0; k < 512; k++) acc += tsr[k] * bf2f(wp[k]);
    gl[(size_t)row * 64 + s] = acc;
}

__global__ __launch_bounds__(256) void nulllogits_k(const float* __restrict__ ns,
                                                    const u16* __restrict__ WvT,
                                                    const float* __restrict__ bv,
                                                    float* __restrict__ nl) {
    __shared__ float s_ns[512];
    for (int i = threadIdx.x; i < 512; i += 256) s_ns[i] = ns[i];
    __syncthreads();
    int col = blockIdx.x * 256 + threadIdx.x;
    const u16* wp = WvT + (size_t)col * 512;
    float acc = bv[col];
    for (int k = 0; k < 512; k++) acc += s_ns[k] * bf2f(wp[k]);
    nl[col] = acc;
}

__global__ __launch_bounds__(1024) void nullreduce_k(const float* __restrict__ nl,
                                                     float* __restrict__ nms) {
    __shared__ float red[1024];
    int tid = threadIdx.x;
    float m = -1e30f;
    for (int i = tid; i < 32000; i += 1024) m = fmaxf(m, nl[i]);
    red[tid] = m; __syncthreads();
    for (int s = 512; s > 0; s >>= 1) { if (tid < s) red[tid] = fmaxf(red[tid], red[tid + s]); __syncthreads(); }
    float gm = red[0]; __syncthreads();
    float sum = 0.f;
    for (int i = tid; i < 32000; i += 1024) sum += expf(nl[i] - gm);
    red[tid] = sum; __syncthreads();
    for (int s = 512; s > 0; s >>= 1) { if (tid < s) red[tid] += red[tid + s]; __syncthreads(); }
    if (tid == 0) { nms[0] = gm; nms[1] = red[0]; }
}

__global__ __launch_bounds__(256) void emis_k(const float* __restrict__ gl,
                                              const float* __restrict__ rowmax,
                                              const float* __restrict__ rowsum,
                                              const float* __restrict__ nl,
                                              const float* __restrict__ nms,
                                              const int* __restrict__ sources,
                                              float* __restrict__ out) {
    int idx = blockIdx.x * 256 + threadIdx.x;  // 16*128*64
    int b = idx >> 13;
    int r = (idx >> 6) & 127;
    int s = idx & 63;
    float v;
    if (r < 64) {
        int row = b * 64 + r;
        v = expf(gl[(size_t)row * 64 + s] - rowmax[row]) / rowsum[row];
    } else {
        int col = sources[b * 64 + s];
        v = expf(nl[col] - nms[0]) / nms[1];
    }
    out[idx] = v;
}

// ---------------------------------------------------------------- transition path tails

__global__ __launch_bounds__(256) void jsoftmax_k(float* __restrict__ jl) {
    __shared__ float red[256];
    int row = blockIdx.x, tid = threadIdx.x;
    float* rp = jl + (size_t)row * 201;
    float m = -1e30f;
    for (int j = tid; j < 201; j += 256) m = fmaxf(m, rp[j]);
    red[tid] = m; __syncthreads();
    for (int s = 128; s > 0; s >>= 1) { if (tid < s) red[tid] = fmaxf(red[tid], red[tid + s]); __syncthreads(); }
    m = red[0]; __syncthreads();
    float sum = 0.f;
    for (int j = tid; j < 201; j += 256) sum += expf(rp[j] - m);
    red[tid] = sum; __syncthreads();
    for (int s = 128; s > 0; s >>= 1) { if (tid < s) red[tid] += red[tid + s]; __syncthreads(); }
    sum = red[0]; __syncthreads();
    float inv = 1.f / sum;
    for (int j = tid; j < 201; j += 256) rp[j] = expf(rp[j] - m) * inv;
}

__global__ __launch_bounds__(256) void p0_k(const float* __restrict__ tt,
                                            const float* __restrict__ Wp0,
                                            const float* __restrict__ bp0,
                                            float* __restrict__ p0) {
    int wv = threadIdx.x >> 5, lane = threadIdx.x & 31;
    int row = blockIdx.x * 8 + wv;
    float acc = 0.f;
    for (int k = lane; k < 512; k += 32) acc += tt[(size_t)row * 512 + k] * Wp0[k];
    for (int d = 16; d >= 1; d >>= 1) acc += __shfl_xor(acc, d, 32);
    if (lane == 0) p0[row] = 1.f / (1.f + expf(-(acc + bp0[0])));
}

__global__ __launch_bounds__(256) void trans_k(const float* __restrict__ jl,
                                               const float* __restrict__ p0,
                                               float* __restrict__ outT,
                                               float* __restrict__ outL) {
    int idx = blockIdx.x * 256 + threadIdx.x;  // 16*128*128
    int b = idx >> 14;
    int i2 = (idx >> 7) & 127;
    int j2 = idx & 127;
    int i = i2 & 63;
    int row = b * 64 + i;
    float v, lv;
    if (j2 < 64) {
        float x = jl[(size_t)row * 201 + 100 + j2 - i];   // band idx in [37,163]
        v = x; lv = logf(x);
    } else {
        int j = j2 - 64;
        if (i == j) { float x = p0[row]; v = x; lv = logf(x); }
        else        { v = 0.f; lv = 0.f; }
    }
    outT[idx] = v;
    outL[idx] = lv;
}

// ---------------------------------------------------------------- launcher

extern "C" void kernel_launch(void* const* d_in, const int* in_sizes, int n_in,
                              void* d_out, int out_size, void* d_ws, size_t ws_size,
                              hipStream_t stream) {
    (void)in_sizes; (void)n_in; (void)out_size; (void)ws_size;
    const int*   sources = (const int*)d_in[0];
    const int*   targets = (const int*)d_in[1];
    const int*   tnull   = (const int*)d_in[2];
    const int*   tlen    = (const int*)d_in[3];
    const float* emb     = (const float*)d_in[4];
    const float* nnW     = (const float*)d_in[5];
    const float* nnb     = (const float*)d_in[6];
    const float* Wx_fw   = (const float*)d_in[7];
    const float* Wh_fw   = (const float*)d_in[8];
    const float* b_fw    = (const float*)d_in[9];
    const float* Wx_bw   = (const float*)d_in[10];
    const float* Wh_bw   = (const float*)d_in[11];
    const float* b_bw    = (const float*)d_in[12];
    const float* Wproj_e = (const float*)d_in[13];
    const float* Wv      = (const float*)d_in[14];
    const float* bv      = (const float*)d_in[15];
    const float* Wproj_t = (const float*)d_in[16];
    const float* Wj      = (const float*)d_in[17];
    const float* bj      = (const float*)d_in[18];
    const float* Wp0     = (const float*)d_in[19];
    const float* bp0     = (const float*)d_in[20];

    float* ws = (float*)d_ws;
    size_t off = 0;
    auto alloc = [&](size_t n) { float* p = ws + off; off += (n + 255) & ~(size_t)255; return p; };
    float*  x      = alloc(262144);    // [1024,256]
    float*  xg_fw  = alloc(2097152);   // [1024,2048]
    float*  xg_bw  = alloc(2097152);
    float*  hcat   = alloc(1048576);   // [1024,1024]
    float*  tsb    = alloc(524288);    // [1024,512]
    float*  tproj  = alloc(262144);    // [1024,256]
    float*  ttb    = alloc(524288);    // [1024,512]
    float*  jl     = alloc(205824);    // [1024,201]
    float*  p0b    = alloc(1024);
    float2* pb     = (float2*)alloc(1024000);  // [1024,500] (max,sum)
    float*  rowmax = alloc(1024);
    float*  rowsum = alloc(1024);
    float*  nstate = alloc(512);
    float*  nlog   = alloc(32000);
    float*  nms    = alloc(2);
    float*  gl     = alloc(65536);     // [1024,64]
    // bf16 transposed weights [NP][K]
    u16* WxTf = (u16*)alloc(262144);   // 2048 x 256
    u16* WxTb = (u16*)alloc(262144);
    u16* WhTf = (u16*)alloc(524288);   // 2048 x 512
    u16* WhTb = (u16*)alloc(524288);
    u16* WpeT = (u16*)alloc(262144);   // 512 x 1024
    u16* WptT = (u16*)alloc(131072);   // 256 x 1024
    u16* nnWT = (u16*)alloc(65536);    // 512 x 256
    u16* WjT  = (u16*)alloc(65536);    // 256(pad) x 512
    u16* WvT  = (u16*)alloc(8192000);  // 32000 x 512

    float* outE = (float*)d_out;       // [16,128,64]
    float* outT = outE + 131072;       // [16,128,128]
    float* outL = outT + 262144;       // [16,128,128]

    // prep: embedding gather + weight transposes to bf16 [N][K]
    embed_k<<<1024, 256, 0, stream>>>(targets, emb, x);
    btr_k<<<2048, 128, 0, stream>>>(Wx_fw, WxTf, 2048, 256);
    btr_k<<<2048, 128, 0, stream>>>(Wx_bw, WxTb, 2048, 256);
    btr_k<<<2048, 256, 0, stream>>>(Wh_fw, WhTf, 2048, 512);
    btr_k<<<2048, 256, 0, stream>>>(Wh_bw, WhTb, 2048, 512);
    btr_k<<<512, 512, 0, stream>>>(Wproj_e, WpeT, 512, 1024);
    btr_k<<<256, 512, 0, stream>>>(Wproj_t, WptT, 256, 1024);
    btr_k<<<512, 128, 0, stream>>>(nnW, nnWT, 512, 256);
    btr_k<<<256, 256, 0, stream>>>(Wj, WjT, 201, 512);
    btr_k<<<32000, 256, 0, stream>>>(Wv, WvT, 32000, 512);
    nullstate_k<<<1, 512, 0, stream>>>(tnull, emb, nnW, nnb, nstate);

    // x @ Wx (+b) for both directions
    gemm_k<0><<<dim3(32, 16), 128, 0, stream>>>(x, WxTf, b_fw, xg_fw, 1024, 2048, 256, 2048);
    gemm_k<0><<<dim3(32, 16), 128, 0, stream>>>(x, WxTb, b_bw, xg_bw, 1024, 2048, 256, 2048);

    // recurrent part, both directions concurrently
    lstm_k<<<2, 512, 0, stream>>>(xg_fw, xg_bw, WhTf, WhTb, tlen, hcat);

    // projections
    gemm_k<0><<<dim3(8, 16), 128, 0, stream>>>(hcat, WpeT, nullptr, tsb, 1024, 512, 1024, 512);
    gemm_k<0><<<dim3(4, 16), 128, 0, stream>>>(hcat, WptT, nullptr, tproj, 1024, 256, 1024, 256);
    gemm_k<1><<<dim3(8, 16), 128, 0, stream>>>(tproj, nnWT, nnb, ttb, 1024, 512, 256, 512);

    // emission: fused logits + softmax stats, then gathered numerators
    logits_reduce_k<<<dim3(500, 16), 128, 0, stream>>>(tsb, WvT, bv, pb);
    combine_k<<<128, 256, 0, stream>>>(pb, rowmax, rowsum);
    gl_k<<<1024, 64, 0, stream>>>(tsb, WvT, bv, sources, gl);
    nulllogits_k<<<125, 256, 0, stream>>>(nstate, WvT, bv, nlog);
    nullreduce_k<<<1, 1024, 0, stream>>>(nlog, nms);
    emis_k<<<512, 256, 0, stream>>>(gl, rowmax, rowsum, nlog, nms, sources, outE);

    // transition
    gemm_k<0><<<dim3(4, 16), 128, 0, stream>>>(ttb, WjT, bj, jl, 1024, 201, 512, 201);
    jsoftmax_k<<<1024, 256, 0, stream>>>(jl);
    p0_k<<<128, 256, 0, stream>>>(ttb, Wp0, bp0, p0b);
    trans_k<<<1024, 256, 0, stream>>>(jl, p0b, outT, outL);
}
